// Attention_87668872446171
// MI455X (gfx1250) — matrix operand order repaired and checked by reading, not compiled, not run on hardware
//
#include <hip/hip_runtime.h>
#include <hip/hip_bf16.h>

// Problem constants (from reference)
#define B_SZ    8
#define SEQ     1024
#define DMODEL  768
#define HEADS   12
#define DHEAD   64
#define QKV_COLS 2304   // 3 * HEADS * DHEAD

typedef __attribute__((ext_vector_type(16))) _Float16 v16h;
typedef __attribute__((ext_vector_type(8)))  _Float16 v8h;
typedef __attribute__((ext_vector_type(8)))  float    v8f;
typedef __attribute__((ext_vector_type(4)))  int      i32x4;

// --- CDNA5 async global->LDS path (ASYNCcnt-tracked) ------------------------
#if defined(__has_builtin)
#if __has_builtin(__builtin_amdgcn_global_load_async_to_lds_b128) && \
    __has_builtin(__builtin_amdgcn_s_wait_asynccnt)
#define USE_ASYNC_LDS 1
#endif
#endif
#ifndef USE_ASYNC_LDS
#define USE_ASYNC_LDS 0
#endif

#define AS_GLOBAL __attribute__((address_space(1)))
#define AS_LDS    __attribute__((address_space(3)))

// Copy 16 bytes from global to LDS. Async path: per-lane
// global_load_async_to_lds_b128 (no VGPR round trip, overlaps with compute).
__device__ __forceinline__ void copy16_g2l(const _Float16* gsrc, _Float16* lds_dst) {
#if USE_ASYNC_LDS
  // Flat shared addresses carry the LDS offset in addr[31:0] (ISA 10.2).
  __builtin_amdgcn_global_load_async_to_lds_b128(
      (AS_GLOBAL i32x4*)(uintptr_t)gsrc,
      (AS_LDS i32x4*)(uint32_t)(uintptr_t)lds_dst, 0, 0);
#else
  *(v8h*)lds_dst = *(const v8h*)gsrc;
#endif
}

__device__ __forceinline__ void g2l_drain() {
#if USE_ASYNC_LDS
  __builtin_amdgcn_s_wait_asynccnt(0);  // drain this wave's ASYNCcnt
#endif
}

// --- WMMA helpers -----------------------------------------------------------
// A-fragment (16x32 f16): as called here, lane holds K = 16*lh..16*lh+7 and
// 16*lh+8..16*lh+15, lh = lane>>4 (contiguous k, as B). Two 16B runs.
__device__ __forceinline__ v16h make_a_frag(const _Float16* lo, const _Float16* hi) {
  v8h a = *(const v8h*)lo;
  v8h b = *(const v8h*)hi;
  v16h r;
#pragma unroll
  for (int i = 0; i < 8; ++i) { r[i] = a[i]; r[i + 8] = b[i]; }
  return r;
}

__device__ __forceinline__ v8f wmma_f16(v16h a, v16h b, v8f c) {
  // (neg_a, A, neg_b, B, c_mod, C, reuse_a, reuse_b)
  return __builtin_amdgcn_wmma_f32_16x16x32_f16(false, a, false, b, (short)0, c,
                                                false, false);
}

// --- Kernel: f32 -> f16 convert --------------------------------------------
__global__ void cvt_f32_f16(const float* __restrict__ src,
                            _Float16* __restrict__ dst, int n) {
  int i = blockIdx.x * 256 + threadIdx.x;
  if (i < n) dst[i] = (_Float16)src[i];
}

// --- Kernel: QKV projection GEMM, scatter to [B,H,N,64] f16 -----------------
// Block: 256 threads (8 waves). Tile: M=128 rows (tokens) x N=64 cols. K=32 steps.
__global__ __launch_bounds__(256) void qkv_gemm_kernel(
    const float* __restrict__ x, const _Float16* __restrict__ wq,
    _Float16* __restrict__ qT, _Float16* __restrict__ kT,
    _Float16* __restrict__ vT) {
  __shared__ __align__(64) _Float16 As[128][32];
  __shared__ __align__(64) _Float16 BsT[64][32];  // [n][k] transposed
  const int m0 = blockIdx.x * 128;
  const int n0 = blockIdx.y * 64;
  const int tid = threadIdx.x;
  const int wave = tid >> 5;
  const int lane = tid & 31;
  const int l15 = lane & 15, lh = lane >> 4, kh = lh * 8;

  v8f acc[4] = {};

  for (int k0 = 0; k0 < DMODEL; k0 += 32) {
    __syncthreads();
    {
      // A tile: 128x32 f32 -> f16; thread loads 16 consecutive floats
      int row = tid >> 1, cb = (tid & 1) * 16;
      const float* src = x + (size_t)(m0 + row) * DMODEL + k0 + cb;
#pragma unroll
      for (int v = 0; v < 4; ++v) {
        float4 f = ((const float4*)src)[v];
        int c = cb + v * 4;
        As[row][c + 0] = (_Float16)f.x;
        As[row][c + 1] = (_Float16)f.y;
        As[row][c + 2] = (_Float16)f.z;
        As[row][c + 3] = (_Float16)f.w;
      }
      // B tile: 32x64 f16, write transposed [n][k]
      int kk = tid >> 3, nb = (tid & 7) * 8;
      v8h wrow = *(const v8h*)(wq + (size_t)(k0 + kk) * QKV_COLS + n0 + nb);
#pragma unroll
      for (int j = 0; j < 8; ++j) BsT[nb + j][kk] = wrow[j];
    }
    __syncthreads();

    const _Float16* arow = &As[wave * 16 + l15][0];
    v16h a = make_a_frag(arow + 16 * lh, arow + 16 * lh + 8);
#pragma unroll
    for (int t = 0; t < 4; ++t) {
      // B-fragment: lane holds K = 16*lh .. 16*lh+15 contiguous
      v16h b = *(const v16h*)&BsT[16 * t + l15][lh * 16];
      acc[t] = wmma_f16(a, b, acc[t]);
    }
  }

  // Epilogue: each 16-col tile lies in exactly one head of q/k/v (n0 % 64 == 0)
#pragma unroll
  for (int t = 0; t < 4; ++t) {
    int c0 = n0 + 16 * t;
    int which = c0 / DMODEL;  // 0=q 1=k 2=v
    int rem = c0 - which * DMODEL;
    int h = rem >> 6;
    int dbase = rem & 63;
    _Float16* dst = (which == 0) ? qT : ((which == 1) ? kT : vT);
#pragma unroll
    for (int r = 0; r < 8; ++r) {
      int bn = m0 + wave * 16 + r + 8 * lh;
      int b = bn >> 10, n = bn & (SEQ - 1);
      dst[(size_t)((b * HEADS + h) * SEQ + n) * DHEAD + dbase + l15] =
          (_Float16)acc[t][r];
    }
  }
}

// --- Kernel: RoPE in place on q,k ([B,H,N,64] f16) --------------------------
__global__ void rope_kernel(_Float16* __restrict__ qT, _Float16* __restrict__ kT) {
  int i = blockIdx.x * 256 + threadIdx.x;  // B*H*N*32 threads
  int d = i & 31;
  int r = i >> 5;           // (b*H + h)*N + n
  int n = r & (SEQ - 1);
  // angle = n * 10000^(-d/32) = n * 2^(-d * log2(1e4)/32)
  float ang = (float)n * exp2f((float)d * (-13.287712379549449f / 32.0f));
  float c = cosf(ang), s = sinf(ang);
  size_t base = (size_t)r * DHEAD;
  float q1 = (float)qT[base + d], q2 = (float)qT[base + d + 32];
  qT[base + d]      = (_Float16)(q1 * c - q2 * s);
  qT[base + d + 32] = (_Float16)(q2 * c + q1 * s);
  float k1 = (float)kT[base + d], k2 = (float)kT[base + d + 32];
  kT[base + d]      = (_Float16)(k1 * c - k2 * s);
  kT[base + d + 32] = (_Float16)(k2 * c + k1 * s);
}

// --- Kernel: flash attention, 64-query tile per block, 4 waves --------------
__global__ __launch_bounds__(128) void flash_attn_kernel(
    const _Float16* __restrict__ qT, const _Float16* __restrict__ kT,
    const _Float16* __restrict__ vT, _Float16* __restrict__ obuf) {
  __shared__ __align__(64) _Float16 Ks[64][64];  // [key][d]
  __shared__ __align__(64) _Float16 Vs[64][64];  // [d][key] (transposed)
  __shared__ __align__(64) _Float16 Ps[64][64];  // [qrow][key]
  const int bh = blockIdx.y;            // b*HEADS + h
  const int q0 = blockIdx.x * 64;
  const int tid = threadIdx.x;
  const int wave = tid >> 5;
  const int lane = tid & 31;
  const int l15 = lane & 15, lh = lane >> 4, kh = lh * 8;
  const size_t head_off = (size_t)bh * SEQ * DHEAD;
  const _Float16* Q = qT + head_off;
  const _Float16* K = kT + head_off;
  const _Float16* V = vT + head_off;

  // Q A-fragments held in registers for the whole kernel (2 K-chunks of 32)
  v16h aQ[2];
  {
    const _Float16* qrow = Q + (size_t)(q0 + wave * 16 + l15) * DHEAD;
#pragma unroll
    for (int c = 0; c < 2; ++c)
      aQ[c] = make_a_frag(qrow + c * 32 + 16 * lh, qrow + c * 32 + 16 * lh + 8);
  }

  float m_i[8], l_i[8];
#pragma unroll
  for (int r = 0; r < 8; ++r) { m_i[r] = -1e30f; l_i[r] = 0.0f; }
  v8f o[4] = {};

  for (int j0 = 0; j0 < SEQ; j0 += 64) {
    __syncthreads();
    {
      // K tile [key][d]: CDNA5 async global->LDS (no VGPR round trip).
      int row = tid >> 1;
      int cb = (tid & 1) * 32;
      const _Float16* ksrc = K + (size_t)(j0 + row) * DHEAD + cb;
#pragma unroll
      for (int v = 0; v < 4; ++v)
        copy16_g2l(ksrc + v * 8, &Ks[row][cb + v * 8]);
      // V tile transposed [d][key]: needs element transpose, manual path.
      const _Float16* vsrc = V + (size_t)(j0 + row) * DHEAD + cb;
#pragma unroll
      for (int v = 0; v < 4; ++v) {
        v8h vv = *(const v8h*)(vsrc + v * 8);
#pragma unroll
        for (int j = 0; j < 8; ++j) Vs[cb + v * 8 + j][row] = vv[j];
      }
    }
    g2l_drain();      // each wave drains its own ASYNCcnt before the barrier
    __syncthreads();  // then the barrier makes all waves' tiles visible

    // S = Q @ K^T for this wave's 16 rows x 64 keys
    v8f s[4] = {};
#pragma unroll
    for (int t = 0; t < 4; ++t) {
#pragma unroll
      for (int c = 0; c < 2; ++c) {
        v16h bK = *(const v16h*)&Ks[16 * t + l15][c * 32 + lh * 16];
        s[t] = wmma_f16(aQ[c], bK, s[t]);
      }
    }

    // Online softmax. Row of elem r = r + 8*lh; its 64 cols live across the
    // 16 lanes of this half in the 4 accumulators -> shfl_xor(1,2,4,8).
    const float scale = 0.125f;  // DHEAD^-0.5
    float mnew[8], lsum[8], alpha[8];
#pragma unroll
    for (int r = 0; r < 8; ++r) {
      float v = fmaxf(fmaxf(s[0][r], s[1][r]), fmaxf(s[2][r], s[3][r]));
#pragma unroll
      for (int m = 1; m < 16; m <<= 1) v = fmaxf(v, __shfl_xor(v, m, 32));
      mnew[r] = fmaxf(m_i[r], v * scale);
      lsum[r] = 0.0f;
    }
#pragma unroll
    for (int t = 0; t < 4; ++t)
#pragma unroll
      for (int r = 0; r < 8; ++r) {
        float p = __expf(s[t][r] * scale - mnew[r]);
        s[t][r] = p;
        lsum[r] += p;
      }
#pragma unroll
    for (int r = 0; r < 8; ++r) {
#pragma unroll
      for (int m = 1; m < 16; m <<= 1) lsum[r] += __shfl_xor(lsum[r], m, 32);
      alpha[r] = __expf(m_i[r] - mnew[r]);
      l_i[r] = l_i[r] * alpha[r] + lsum[r];
      m_i[r] = mnew[r];
    }
#pragma unroll
    for (int t = 0; t < 4; ++t)
#pragma unroll
      for (int r = 0; r < 8; ++r) o[t][r] *= alpha[r];

    // Spill P (f16) to wave-private LDS rows to re-read in A-fragment layout.
    // Same-wave LDS ops are in-order; no cross-wave sharing of these rows.
#pragma unroll
    for (int t = 0; t < 4; ++t)
#pragma unroll
      for (int r = 0; r < 8; ++r)
        Ps[wave * 16 + r + 8 * lh][16 * t + l15] = (_Float16)s[t][r];

    // O += P @ V
#pragma unroll
    for (int c = 0; c < 2; ++c) {
      const _Float16* prow = &Ps[wave * 16 + l15][0];
      v16h aP = make_a_frag(prow + c * 32 + 16 * lh, prow + c * 32 + 16 * lh + 8);
#pragma unroll
      for (int t = 0; t < 4; ++t) {
        v16h bV = *(const v16h*)&Vs[16 * t + l15][c * 32 + lh * 16];
        o[t] = wmma_f16(aP, bV, o[t]);
      }
    }
  }

  // Normalize and store to [B, N, H*64] f16
  const int b = bh / HEADS;
  const int h = bh - b * HEADS;
#pragma unroll
  for (int t = 0; t < 4; ++t)
#pragma unroll
    for (int r = 0; r < 8; ++r) {
      int n = q0 + wave * 16 + r + 8 * lh;
      int col = h * DHEAD + 16 * t + l15;
      obuf[(size_t)(b * SEQ + n) * DMODEL + col] = (_Float16)(o[t][r] / l_i[r]);
    }
}

// --- Kernel: output projection GEMM + bias, f32 out -------------------------
__global__ __launch_bounds__(256) void out_gemm_kernel(
    const _Float16* __restrict__ A, const _Float16* __restrict__ w,
    const float* __restrict__ bias, float* __restrict__ out) {
  __shared__ __align__(64) _Float16 As[128][32];
  __shared__ __align__(64) _Float16 BsT[64][32];
  const int m0 = blockIdx.x * 128;
  const int n0 = blockIdx.y * 64;
  const int tid = threadIdx.x;
  const int wave = tid >> 5;
  const int lane = tid & 31;
  const int l15 = lane & 15, lh = lane >> 4, kh = lh * 8;

  v8f acc[4] = {};

  for (int k0 = 0; k0 < DMODEL; k0 += 32) {
    __syncthreads();
    {
      // A tile (already f16, row-major copy): async global->LDS.
      int row = tid >> 1, cb = (tid & 1) * 16;
      const _Float16* src = A + (size_t)(m0 + row) * DMODEL + k0 + cb;
      copy16_g2l(src,     &As[row][cb]);
      copy16_g2l(src + 8, &As[row][cb + 8]);
      // B tile: 32x64 f16, write transposed [n][k] (manual).
      int kk = tid >> 3, nb = (tid & 7) * 8;
      v8h wrow = *(const v8h*)(w + (size_t)(k0 + kk) * DMODEL + n0 + nb);
#pragma unroll
      for (int j = 0; j < 8; ++j) BsT[nb + j][kk] = wrow[j];
    }
    g2l_drain();
    __syncthreads();

    const _Float16* arow = &As[wave * 16 + l15][0];
    v16h a = make_a_frag(arow + 16 * lh, arow + 16 * lh + 8);
#pragma unroll
    for (int t = 0; t < 4; ++t) {
      v16h b = *(const v16h*)&BsT[16 * t + l15][lh * 16];
      acc[t] = wmma_f16(a, b, acc[t]);
    }
  }

#pragma unroll
  for (int t = 0; t < 4; ++t) {
    int c = n0 + 16 * t + l15;
    float bv = bias[c];
#pragma unroll
    for (int r = 0; r < 8; ++r) {
      int row = m0 + wave * 16 + r + 8 * lh;
      out[(size_t)row * DMODEL + c] = acc[t][r] + bv;
    }
  }
}

// --- Launch -----------------------------------------------------------------
extern "C" void kernel_launch(void* const* d_in, const int* in_sizes, int n_in,
                              void* d_out, int out_size, void* d_ws, size_t ws_size,
                              hipStream_t stream) {
  const float* x     = (const float*)d_in[0];  // [8,1024,768]
  const float* w_qkv = (const float*)d_in[1];  // [768,2304]
  const float* w_out = (const float*)d_in[2];  // [768,768]
  const float* b_out = (const float*)d_in[3];  // [768]
  float* out = (float*)d_out;                  // [8,1024,768] f32

  // Workspace layout (halves): ~53 MB total
  _Float16* ws      = (_Float16*)d_ws;
  _Float16* w_qkv_h = ws;                                   // 768*2304
  _Float16* w_out_h = w_qkv_h + 768 * 2304;                 // 768*768
  _Float16* qT      = w_out_h + 768 * 768;                  // B*H*N*64
  _Float16* kT      = qT + (size_t)B_SZ * HEADS * SEQ * DHEAD;
  _Float16* vT      = kT + (size_t)B_SZ * HEADS * SEQ * DHEAD;
  _Float16* obuf    = vT + (size_t)B_SZ * HEADS * SEQ * DHEAD;  // B*N*768

  cvt_f32_f16<<<(768 * 2304 + 255) / 256, 256, 0, stream>>>(w_qkv, w_qkv_h,
                                                            768 * 2304);
  cvt_f32_f16<<<(768 * 768 + 255) / 256, 256, 0, stream>>>(w_out, w_out_h,
                                                           768 * 768);
  qkv_gemm_kernel<<<dim3(64, 36), 256, 0, stream>>>(x, w_qkv_h, qT, kT, vT);
  rope_kernel<<<(B_SZ * HEADS * SEQ * 32) / 256, 256, 0, stream>>>(qT, kT);
  flash_attn_kernel<<<dim3(SEQ / 64, B_SZ * HEADS), 128, 0, stream>>>(qT, kT, vT,
                                                                      obuf);
  out_gemm_kernel<<<dim3(64, 12), 256, 0, stream>>>(obuf, w_out_h, b_out, out);
}
